// ShenzhenGeometryWithFBP_74062416053310
// MI455X (gfx1250) — compile-verified, matrix operand order checked by reading, not run on hardware
//
#include <hip/hip_runtime.h>

// ---------------- geometry constants ----------------
#define NUM_ANG 96      // A
#define DET_W   225     // W
#define DET_H   24      // H (== Z)
#define NV      160     // N (volume X=Y)
#define NZ      24      // Z slices
#define NSAMP   224     // T ray samples
#define KPAD    256     // W padded to multiple of 32 (WMMA K)
#define NPAD    240     // W padded to multiple of 16 (WMMA N)
#define NROWS   (NUM_ANG * DET_H)   // 2304 sinogram rows (multiple of 16)
#define PI_F    3.14159265358979323846f

typedef __attribute__((ext_vector_type(16))) _Float16 v16h;
typedef __attribute__((ext_vector_type(8)))  _Float16 v8h;
typedef __attribute__((ext_vector_type(8)))  float    v8f;
typedef __attribute__((ext_vector_type(4)))  int      v4i;

#define AS1 __attribute__((address_space(1)))
#define AS3 __attribute__((address_space(3)))

// ---------------------------------------------------------------------------
// K1: ray-driven forward projection, residual, cos-weight; emit f16 rows
// padded to KPAD with zeros (ready as WMMA A-matrix source).
// grid = NROWS blocks, block = 256 threads (threadIdx.x = padded column).
// ---------------------------------------------------------------------------
__global__ void fp_residual_kernel(const float* __restrict__ vol,
                                   const float* __restrict__ p,
                                   _Float16* __restrict__ Rf16) {
    const int row = blockIdx.x;            // 0..2303  (= a*24 + z)
    const int u   = threadIdx.x;           // 0..255
    if (u >= DET_W) {                      // K-padding columns
        Rf16[row * KPAD + u] = (_Float16)0.0f;
        return;
    }
    const int a = row / DET_H;
    const int z = row - a * DET_H;

    float st, ct;
    __sincosf((float)a * (PI_F / (float)NUM_ANG), &st, &ct);

    const float uc = (float)u - 112.0f;               // u - (W-1)/2
    const float* __restrict__ sl = vol + z * NV * NV; // slice [160][160]

    float sum = 0.0f;
    for (int t = 0; t < NSAMP; ++t) {
        float tc = (float)t - 111.5f;                 // t - (T-1)/2
        float xs = fmaf(uc, ct, fmaf(-tc, st, 79.5f));
        float ys = fmaf(uc, st, fmaf( tc, ct, 79.5f));
        float fx = floorf(xs), fy = floorf(ys);
        int   ix = (int)fx,    iy = (int)fy;
        float dx = xs - fx,    dy = ys - fy;
        bool x0 = ((unsigned)ix       < (unsigned)NV);
        bool x1 = ((unsigned)(ix + 1) < (unsigned)NV);
        bool y0 = ((unsigned)iy       < (unsigned)NV);
        bool y1 = ((unsigned)(iy + 1) < (unsigned)NV);
        const float* r0 = sl + iy * NV;
        float v00 = (x0 && y0) ? r0[ix]          : 0.0f;
        float v01 = (x1 && y0) ? r0[ix + 1]      : 0.0f;
        float v10 = (x0 && y1) ? r0[NV + ix]     : 0.0f;
        float v11 = (x1 && y1) ? r0[NV + ix + 1] : 0.0f;
        float top = fmaf(dx, v01 - v00, v00);
        float bot = fmaf(dx, v11 - v10, v10);
        sum += fmaf(dy, bot - top, top);
    }

    float zz = (float)z - 11.5f;                      // z - (Z-1)/2
    float cw = 905.0f / sqrtf(905.0f * 905.0f + uc * uc + zz * zz);
    float res = (sum - p[row * DET_W + u]) * cw;
    Rf16[row * KPAD + u] = (_Float16)res;
}

// ---------------------------------------------------------------------------
// K2: banded ramp-filter matrix, stored column-major for the B operand:
// Gt[n][k] = f[k - n + MID]  (zero outside band / in padding).
// grid = NPAD blocks, block = 256 (threadIdx.x = k).
// ---------------------------------------------------------------------------
__global__ void build_filter_kernel(const float* __restrict__ ramp,
                                    _Float16* __restrict__ Gt) {
    const int n = blockIdx.x;      // output position (GEMM N), 0..239
    const int k = threadIdx.x;     // input position  (GEMM K), 0..255
    const int j = k - n + 112;
    float val = 0.0f;
    if (k < DET_W && n < DET_W && j >= 0 && j < DET_W) val = ramp[j];
    Gt[n * KPAD + k] = (_Float16)val;
}

// ---------------------------------------------------------------------------
// K3: ramp filtering as WMMA GEMM: filt[2304x225] = Rf16[2304x256] x Gt^T.
// Block = 8 waves = 8 consecutive M tiles sharing ONE N tile, so the 8 KB
// B tile is staged once per block into LDS with the CDNA5 async global->LDS
// path (ASYNCcnt), then fed to 8 x v_wmma_f32_16x16x32_f16 per wave.
// A/B lane layout (16-bit, 16x32): lanes 0-15 hold K {0..7,16..23},
// lanes 16-31 hold K {8..15,24..31} -> two contiguous 16-byte loads each.
// ---------------------------------------------------------------------------
__global__ void ramp_gemm_kernel(const _Float16* __restrict__ Rf16,
                                 const _Float16* __restrict__ Gt,
                                 float* __restrict__ filt) {
    __shared__ _Float16 Bsh[16 * KPAD];           // 8192 bytes: one B tile

    const int tid  = threadIdx.x;
    const int lane = tid & 31;
    const int wave = tid >> 5;                    // 0..7
    const int nt   = blockIdx.x % (NPAD / 16);    // N tile (shared by block)
    const int mg   = blockIdx.x / (NPAD / 16);    // M tile group (0..17)
    const int mt   = mg * 8 + wave;               // this wave's M tile (<144)

    // ---- stage B tile (Gt rows nt*16 .. nt*16+15) into LDS: 32 B/thread ----
    {
        const char* gsrc = (const char*)(Gt + (size_t)nt * 16 * KPAD) + tid * 16;
        char*       ldst = (char*)Bsh + tid * 16;
#if __has_builtin(__builtin_amdgcn_global_load_async_to_lds_b128)
        __builtin_amdgcn_global_load_async_to_lds_b128(
            (AS1 v4i*)gsrc,          (AS3 v4i*)ldst,          0, 0);
        __builtin_amdgcn_global_load_async_to_lds_b128(
            (AS1 v4i*)(gsrc + 4096), (AS3 v4i*)(ldst + 4096), 0, 0);
#if __has_builtin(__builtin_amdgcn_s_wait_asynccnt)
        __builtin_amdgcn_s_wait_asynccnt(0);
#else
        asm volatile("s_wait_asynccnt 0x0" ::: "memory");
#endif
#else
        // fallback: synchronous stage via VGPRs (still LDS-shared B tile)
        v8h t0 = *(const v8h*)gsrc;
        v8h t1 = *(const v8h*)(gsrc + 4096);
        *(v8h*)ldst = t0;
        *(v8h*)(ldst + 4096) = t1;
#endif
    }
    __syncthreads();

    const int laneLo = lane & 15;
    const int koff   = (lane >> 4) << 3;   // 0 for lanes 0-15, 8 for 16-31
    const _Float16* __restrict__ arow = Rf16 + (mt * 16 + laneLo) * KPAD + koff;
    const _Float16*              brow = Bsh  + laneLo * KPAD + koff;   // LDS

    v8f c = {};
    #pragma unroll
    for (int kc = 0; kc < KPAD / 32; ++kc) {
        const int kb = kc * 32;
        v8h alo = *(const v8h*)(arow + kb);        // K = kb+koff .. +7
        v8h ahi = *(const v8h*)(arow + kb + 16);   // K = kb+koff+16 .. +23
        v8h blo = *(const v8h*)(brow + kb);        // ds_load_b128
        v8h bhi = *(const v8h*)(brow + kb + 16);
        v16h A = __builtin_shufflevector(alo, ahi, 0,1,2,3,4,5,6,7,8,9,10,11,12,13,14,15);
        v16h B = __builtin_shufflevector(blo, bhi, 0,1,2,3,4,5,6,7,8,9,10,11,12,13,14,15);
        // (neg_a, A, neg_b, B, c_mod, C, reuse_a, reuse_b)
        c = __builtin_amdgcn_wmma_f32_16x16x32_f16(false, A, false, B,
                                                   (short)0, c, false, false);
    }

    // C/D layout: VGPR r -> M = r + 8*(lane>=16), N = lane & 15
    const int col = nt * 16 + laneLo;
    if (col < DET_W) {
        const int rbase = mt * 16 + ((lane >> 4) << 3);
        #pragma unroll
        for (int r = 0; r < 8; ++r)
            filt[(rbase + r) * DET_W + col] = c[r];
    }
}

// ---------------------------------------------------------------------------
// K4: voxel-driven backprojection + final combine: out = x - lamb * BP(filt).
// grid = Z*N*N/256 blocks; per-block cos/sin table in LDS; prefetch next
// angle's sinogram row (L2-resident) ahead of the dependent lerp loads.
// ---------------------------------------------------------------------------
__global__ void bp_combine_kernel(const float* __restrict__ xin,
                                  const float* __restrict__ filt,
                                  const float* __restrict__ lamb,
                                  float* __restrict__ out) {
    __shared__ float cA[NUM_ANG], sA[NUM_ANG];
    const int tid = threadIdx.x;
    if (tid < NUM_ANG) {
        float s, cc;
        __sincosf((float)tid * (PI_F / (float)NUM_ANG), &s, &cc);
        cA[tid] = cc;
        sA[tid] = s;
    }
    __syncthreads();

    const int idx = blockIdx.x * blockDim.x + tid;   // < 614400
    const int x = idx % NV;
    const int y = (idx / NV) % NV;
    const int z = idx / (NV * NV);
    const float xc = (float)x - 79.5f;
    const float yc = (float)y - 79.5f;

    float acc = 0.0f;
    for (int aIdx = 0; aIdx < NUM_ANG; ++aIdx) {
        const float* __restrict__ prow = filt + (aIdx * DET_H + z) * DET_W;
        // speculative prefetch of the next angle's row (global_prefetch_b8)
        __builtin_prefetch(prow + DET_H * DET_W, 0, 1);
        float u  = fmaf(xc, cA[aIdx], fmaf(yc, sA[aIdx], 112.0f));
        float fu = floorf(u);
        int   i0 = (int)fu;
        float w  = u - fu;
        float v0 = ((unsigned)i0       < (unsigned)DET_W) ? prow[i0]     : 0.0f;
        float v1 = ((unsigned)(i0 + 1) < (unsigned)DET_W) ? prow[i0 + 1] : 0.0f;
        acc += fmaf(w, v1 - v0, v0);
    }
    out[idx] = xin[idx] - lamb[0] * acc;
}

// ---------------------------------------------------------------------------
extern "C" void kernel_launch(void* const* d_in, const int* in_sizes, int n_in,
                              void* d_out, int out_size, void* d_ws, size_t ws_size,
                              hipStream_t stream) {
    const float* x    = (const float*)d_in[0];   // [1,1,24,160,160]
    const float* p    = (const float*)d_in[1];   // [1,1,96,24,225]
    const float* ramp = (const float*)d_in[2];   // [1,1,1,225]
    const float* lamb = (const float*)d_in[3];   // scalar
    float* out = (float*)d_out;                  // [1,1,24,160,160]

    char* ws = (char*)d_ws;
    const size_t szR = (size_t)NROWS * KPAD * sizeof(_Float16);  // 1,179,648 B
    const size_t szG = (size_t)NPAD  * KPAD * sizeof(_Float16);  //   122,880 B
    _Float16* Rf16 = (_Float16*)ws;
    _Float16* Gt   = (_Float16*)(ws + szR);
    float*    filt = (float*)(ws + szR + szG);                   // 2,073,600 B

    // 1) forward projection + residual + cos weighting (f16, K-padded)
    fp_residual_kernel<<<NROWS, 256, 0, stream>>>(x, p, Rf16);
    // 2) banded filter matrix (B operand, column-major)
    build_filter_kernel<<<NPAD, 256, 0, stream>>>(ramp, Gt);
    // 3) ramp filtering as WMMA GEMM: 18 M-groups x 15 N tiles -> 270 blocks
    ramp_gemm_kernel<<<(NROWS / 16 / 8) * (NPAD / 16), 256, 0, stream>>>(Rf16, Gt, filt);
    // 4) backprojection + x - lamb*bp
    bp_combine_kernel<<<(NZ * NV * NV) / 256, 256, 0, stream>>>(x, filt, lamb, out);
}
